// HymbaSelfAttention_75746043232514
// MI455X (gfx1250) — matrix operand order repaired and checked by reading, not compiled, not run on hardware
//
#include <hip/hip_runtime.h>
#include <hip/hip_bf16.h>

typedef __bf16 bf16;
typedef bf16 bf16x8  __attribute__((ext_vector_type(8)));
typedef bf16 bf16x16 __attribute__((ext_vector_type(16)));
typedef float f32x8  __attribute__((ext_vector_type(8)));

#define T_SEQ   3072
#define E_DIM   768
#define NH      12
#define NKV     3
#define HD      64
#define WIN     1024
#define QTILES  (T_SEQ / 16)          // 192

__device__ __forceinline__ int imin(int a, int b) { return a < b ? a : b; }
__device__ __forceinline__ int imax(int a, int b) { return a > b ? a : b; }

// A-fragment (16x32 bf16): lane row m = lane%16; klo = 8*(lane/16).
// Elements 0..15 -> K = 2*klo..2*klo+15 (contiguous, matches B).
__device__ __forceinline__ bf16x16 load_a_frag(const bf16* row_base, int klo) {
  bf16x8 lo = *(const bf16x8*)(row_base + 2 * klo);
  bf16x8 hi = *(const bf16x8*)(row_base + 2 * klo + 8);
  bf16x16 r;
#pragma unroll
  for (int i = 0; i < 8; ++i) { r[i] = lo[i]; r[i + 8] = hi[i]; }
  return r;
}

// B-fragment (32x16 bf16): lane column N = lane%16; klo = 16*(lane/16);
// contiguous 16 K-values per lane (32-byte load from transposed operand row).
__device__ __forceinline__ bf16x16 load_b_frag(const bf16* col_base, int klo) {
  return *(const bf16x16*)(col_base + klo);
}

__device__ __forceinline__ f32x8 wmma_bf16(bf16x16 a, bf16x16 b, f32x8 c) {
  return __builtin_amdgcn_wmma_f32_16x16x32_bf16(false, a, false, b, (short)0, c,
                                                 false, false);
}

// ---------------------------------------------------------------- prep casts
__global__ void cast_f32_to_bf16(const float* __restrict__ in,
                                 bf16* __restrict__ out, int n) {
  int i = blockIdx.x * blockDim.x + threadIdx.x;
  if (i < n) out[i] = (bf16)in[i];
}

// in: [R][C] f32 row-major  ->  out: [C][R] bf16 (transposed)
__global__ void transpose_cast_bf16(const float* __restrict__ in,
                                    bf16* __restrict__ out, int R, int C) {
  int i = blockIdx.x * blockDim.x + threadIdx.x;
  if (i < R * C) {
    int r = i / C, c = i % C;
    out[(size_t)c * R + r] = (bf16)in[i];
  }
}

// ------------------------------------------------- fused QKV GEMM + RoPE
// One wave computes one 16-token x 64-dim head tile of Q, K, or V.
// Q/K get RoPE; V is stored transposed [kvh][d][T] for the P@V GEMM.
__global__ __launch_bounds__(256) void qkv_rope_kernel(
    const bf16* __restrict__ xb, const bf16* __restrict__ WqT,
    const bf16* __restrict__ WkT, const bf16* __restrict__ WvT,
    bf16* __restrict__ qh, bf16* __restrict__ kh, bf16* __restrict__ vtp) {
  const int wave = (blockIdx.x * 256 + threadIdx.x) >> 5;
  const int lane = threadIdx.x & 31;
  const int NTQ = NH * QTILES, NTKV = NKV * QTILES;
  if (wave >= NTQ + 2 * NTKV) return;

  int kind, head, tb, w = wave;
  if (w < NTQ) { kind = 0; head = w / QTILES; tb = w % QTILES; }
  else {
    w -= NTQ;
    if (w < NTKV) kind = 1; else { kind = 2; w -= NTKV; }
    head = w / QTILES; tb = w % QTILES;
  }
  const bf16* Wt = (kind == 0) ? WqT : (kind == 1) ? WkT : WvT;

  const int t0 = tb * 16;
  const int half = lane >> 4, col = lane & 15;
  const int kloA = half * 8, kloB = half * 16;

  const bf16* arow = xb + (size_t)(t0 + col) * E_DIM;   // A row = token
  const bf16* wrow[4];
#pragma unroll
  for (int nt = 0; nt < 4; ++nt)
    wrow[nt] = Wt + (size_t)(head * HD + nt * 16 + col) * E_DIM;  // B col = outdim

  f32x8 acc[4] = {};
  for (int kb = 0; kb < E_DIM; kb += 32) {
    bf16x16 a = load_a_frag(arow + kb, kloA);
#pragma unroll
    for (int nt = 0; nt < 4; ++nt) {
      bf16x16 b = load_b_frag(wrow[nt] + kb, kloB);
      acc[nt] = wmma_bf16(a, b, acc[nt]);
    }
  }

  if (kind != 2) {  // RoPE: pair (d, d+32) lives in acc[nt] / acc[nt+2], same slot
#pragma unroll
    for (int nt = 0; nt < 2; ++nt) {
      int d = nt * 16 + col;                        // 0..31
      float invf = __expf(-(float)d * 0.28782313662427283f);  // 10000^(-d/32)
#pragma unroll
      for (int r = 0; r < 8; ++r) {
        int t = t0 + r + 8 * half;
        float fr = (float)t * invf;
        float s = sinf(fr), c = cosf(fr);
        float x1 = acc[nt][r], x2 = acc[nt + 2][r];
        acc[nt][r]     = x1 * c + x2 * s;
        acc[nt + 2][r] = -x1 * s + x2 * c;
      }
    }
  }

  if (kind == 2) {  // V transposed: vtp[kvh][d][t]
#pragma unroll
    for (int nt = 0; nt < 4; ++nt) {
      bf16* base = vtp + (size_t)(head * HD + nt * 16 + col) * T_SEQ;
#pragma unroll
      for (int r = 0; r < 8; ++r) base[t0 + r + 8 * half] = (bf16)acc[nt][r];
    }
  } else {          // Q/K head-major: [head][t][d]
    bf16* dst = (kind == 0 ? qh : kh) + (size_t)head * T_SEQ * HD;
#pragma unroll
    for (int nt = 0; nt < 4; ++nt) {
      int d = nt * 16 + col;
#pragma unroll
      for (int r = 0; r < 8; ++r)
        dst[(size_t)(t0 + r + 8 * half) * HD + d] = (bf16)acc[nt][r];
    }
  }
}

// ------------------------------------------------- sliding-window flash attn
__global__ __launch_bounds__(256) void attn_kernel(
    const bf16* __restrict__ qh, const bf16* __restrict__ kh,
    const bf16* __restrict__ vtp, bf16* __restrict__ att) {
  __shared__ bf16 Plds[8][16][32];  // per-wave P staging (C/D -> A layout)
  const int wib = threadIdx.x >> 5;
  const int wave = (blockIdx.x * 256 + threadIdx.x) >> 5;
  const int lane = threadIdx.x & 31;
  if (wave >= NH * QTILES) return;
  const int head = wave / QTILES, qb = wave % QTILES;
  const int kvh = head >> 2;      // KV_REPEAT = 4
  const int t0 = qb * 16;
  const int half = lane >> 4, col = lane & 15;
  const int kloA = half * 8, kloB = half * 16;
  const float SCALE_F = 0.21650635094610965f;  // (HD*4/12)^-0.5

  const bf16* qrow = qh + ((size_t)head * T_SEQ + t0 + col) * HD;
  bf16x16 qa0 = load_a_frag(qrow, kloA);        // d 0..31
  bf16x16 qa1 = load_a_frag(qrow + 32, kloA);   // d 32..63

  const bf16* kbase = kh + (size_t)kvh * T_SEQ * HD;
  const bf16* vbase = vtp + (size_t)kvh * HD * T_SEQ;

  float m_i[8], l_i[8];
#pragma unroll
  for (int r = 0; r < 8; ++r) { m_i[r] = -1e30f; l_i[r] = 0.0f; }
  f32x8 o[4] = {};

  const int smin = imin(imax(t0 - WIN / 2, 0), T_SEQ - WIN);
  const int smax = imin(imax(t0 + 15 - WIN / 2, 0), T_SEQ - WIN);
  const int kbeg = smin & ~31;
  const int kend = imin(T_SEQ, (smax + WIN + 31) & ~31);

  for (int k0 = kbeg; k0 < kend; k0 += 32) {
    if (k0 + 32 < kend) {  // warm next chunk (global_prefetch_b8)
      __builtin_prefetch(kbase + (size_t)(k0 + 32 + col) * HD, 0, 3);
      __builtin_prefetch(vbase + (size_t)col * T_SEQ + k0 + 32, 0, 3);
    }
    f32x8 s[2];
#pragma unroll
    for (int kt = 0; kt < 2; ++kt) {
      const bf16* krow = kbase + (size_t)(k0 + kt * 16 + col) * HD;  // B col = key
      f32x8 sc = {};
      sc = wmma_bf16(qa0, load_b_frag(krow, kloB), sc);
      sc = wmma_bf16(qa1, load_b_frag(krow + 32, kloB), sc);
#pragma unroll
      for (int r = 0; r < 8; ++r) {  // scale + sliding-window mask
        int i = t0 + r + 8 * half;
        int j = k0 + kt * 16 + col;
        int st = imin(imax(i - WIN / 2, 0), T_SEQ - WIN);
        bool valid = (j >= st) && (j < st + WIN);
        sc[r] = valid ? sc[r] * SCALE_F : -1e30f;
      }
      s[kt] = sc;
    }
    // online softmax over this 32-key chunk
#pragma unroll
    for (int r = 0; r < 8; ++r) {
      float rm = fmaxf(s[0][r], s[1][r]);
#pragma unroll
      for (int off = 1; off <= 8; off <<= 1)
        rm = fmaxf(rm, __shfl_xor(rm, off, 32));
      float mn = fmaxf(m_i[r], rm);
      float resc = __expf(m_i[r] - mn);
      float p0 = __expf(s[0][r] - mn);
      float p1 = __expf(s[1][r] - mn);
      float rs = p0 + p1;
#pragma unroll
      for (int off = 1; off <= 8; off <<= 1) rs += __shfl_xor(rs, off, 32);
      l_i[r] = l_i[r] * resc + rs;
      m_i[r] = mn;
#pragma unroll
      for (int nt = 0; nt < 4; ++nt) o[nt][r] *= resc;
      Plds[wib][r + 8 * half][col]      = (bf16)p0;   // C/D layout -> LDS
      Plds[wib][r + 8 * half][16 + col] = (bf16)p1;
    }
    asm volatile("s_wait_dscnt 0" ::: "memory");       // cross-lane DS WAR/RAW
    bf16x16 pa = load_a_frag(&Plds[wib][col][0], kloA);  // A-layout reload
#pragma unroll
    for (int nt = 0; nt < 4; ++nt) {
      const bf16* vcol = vbase + (size_t)(nt * 16 + col) * T_SEQ + k0;
      o[nt] = wmma_bf16(pa, load_b_frag(vcol, kloB), o[nt]);
    }
  }
  // normalize and store token-major [t][head*64+d] for the output GEMM
#pragma unroll
  for (int nt = 0; nt < 4; ++nt) {
    int d = head * HD + nt * 16 + col;
#pragma unroll
    for (int r = 0; r < 8; ++r) {
      int t = t0 + r + 8 * half;
      att[(size_t)t * E_DIM + d] = (bf16)(o[nt][r] / l_i[r]);
    }
  }
}

// ------------------------------------------------- output projection (f32 out)
__global__ __launch_bounds__(256) void out_proj_kernel(
    const bf16* __restrict__ att, const bf16* __restrict__ WoT,
    float* __restrict__ out) {
  const int wave = (blockIdx.x * 256 + threadIdx.x) >> 5;
  const int lane = threadIdx.x & 31;
  if (wave >= QTILES * (E_DIM / HD)) return;
  const int tb = wave / (E_DIM / HD), nb = wave % (E_DIM / HD);
  const int t0 = tb * 16;
  const int half = lane >> 4, col = lane & 15;
  const int kloA = half * 8, kloB = half * 16;

  const bf16* arow = att + (size_t)(t0 + col) * E_DIM;
  const bf16* wrow[4];
#pragma unroll
  for (int nt = 0; nt < 4; ++nt)
    wrow[nt] = WoT + (size_t)(nb * HD + nt * 16 + col) * E_DIM;

  f32x8 acc[4] = {};
  for (int kb = 0; kb < E_DIM; kb += 32) {
    bf16x16 a = load_a_frag(arow + kb, kloA);
#pragma unroll
    for (int nt = 0; nt < 4; ++nt)
      acc[nt] = wmma_bf16(a, load_b_frag(wrow[nt] + kb, kloB), acc[nt]);
  }
#pragma unroll
  for (int nt = 0; nt < 4; ++nt) {
    int n = nb * HD + nt * 16 + col;
#pragma unroll
    for (int r = 0; r < 8; ++r)
      out[(size_t)(t0 + r + 8 * half) * E_DIM + n] = acc[nt][r];
  }
}

// ---------------------------------------------------------------- launcher
extern "C" void kernel_launch(void* const* d_in, const int* in_sizes, int n_in,
                              void* d_out, int out_size, void* d_ws,
                              size_t ws_size, hipStream_t stream) {
  const float* x  = (const float*)d_in[0];
  const float* Wq = (const float*)d_in[1];
  const float* Wk = (const float*)d_in[2];
  const float* Wv = (const float*)d_in[3];
  const float* Wo = (const float*)d_in[4];

  char* ws = (char*)d_ws;
  bf16* xb  = (bf16*)ws;                 ws += (size_t)T_SEQ * E_DIM * 2;
  bf16* WqT = (bf16*)ws;                 ws += (size_t)E_DIM * (NH * HD) * 2;
  bf16* WkT = (bf16*)ws;                 ws += (size_t)E_DIM * (NKV * HD) * 2;
  bf16* WvT = (bf16*)ws;                 ws += (size_t)E_DIM * (NKV * HD) * 2;
  bf16* WoT = (bf16*)ws;                 ws += (size_t)(NH * HD) * E_DIM * 2;
  bf16* qh  = (bf16*)ws;                 ws += (size_t)NH * T_SEQ * HD * 2;
  bf16* kh  = (bf16*)ws;                 ws += (size_t)NKV * T_SEQ * HD * 2;
  bf16* vtp = (bf16*)ws;                 ws += (size_t)NKV * HD * T_SEQ * 2;
  bf16* att = (bf16*)ws;                 ws += (size_t)T_SEQ * E_DIM * 2;

  const int nX = T_SEQ * E_DIM;
  cast_f32_to_bf16<<<(nX + 255) / 256, 256, 0, stream>>>(x, xb, nX);
  transpose_cast_bf16<<<(E_DIM * NH * HD + 255) / 256, 256, 0, stream>>>(
      Wq, WqT, E_DIM, NH * HD);
  transpose_cast_bf16<<<(E_DIM * NKV * HD + 255) / 256, 256, 0, stream>>>(
      Wk, WkT, E_DIM, NKV * HD);
  transpose_cast_bf16<<<(E_DIM * NKV * HD + 255) / 256, 256, 0, stream>>>(
      Wv, WvT, E_DIM, NKV * HD);
  transpose_cast_bf16<<<(NH * HD * E_DIM + 255) / 256, 256, 0, stream>>>(
      Wo, WoT, NH * HD, E_DIM);

  // (12 + 3 + 3) head-slots * 192 token tiles = 3456 waves -> 432 blocks
  qkv_rope_kernel<<<432, 256, 0, stream>>>(xb, WqT, WkT, WvT, qh, kh, vtp);
  // 12 heads * 192 query tiles = 2304 waves -> 288 blocks
  attn_kernel<<<288, 256, 0, stream>>>(qh, kh, vtp, att);
  // 192 * 12 output tiles = 2304 waves -> 288 blocks
  out_proj_kernel<<<288, 256, 0, stream>>>(att, WoT, (float*)d_out);
}
